// TokenizationLayer_86930138071098
// MI455X (gfx1250) — 8-bit matrix operand order repaired and checked by reading, compiled, not run on hardware
//
#include <hip/hip_runtime.h>

typedef __attribute__((ext_vector_type(8))) int v8i;

#define B_ 32
#define C_ 32
#define T_ 8192
#define N_ 256
#define P_ 6
#define TVALID (T_ - P_ + 1)


__global__ __launch_bounds__(256) void prep_patterns(const float* __restrict__ pat,
                                                     unsigned int* __restrict__ Bws,
                                                     int* __restrict__ psum) {
  __shared__ float maxd[P_ * N_];
  __shared__ int   pcnt[N_];
  const int tid = threadIdx.x;

  for (int n = tid; n < N_; n += 256) pcnt[n] = 0;
  __syncthreads();

  for (int idx = tid; idx < P_ * N_; idx += 256) {
    const int n = idx & (N_ - 1);
    const int p = idx >> 8;
    float mx = -__builtin_inff();
    float sm = 0.0f;
    for (int c = 0; c < C_; ++c) {
      const float v = pat[(c * P_ + p) * N_ + n];
      mx = fmaxf(mx, v);
      sm += v;
    }
    int cnt = 0;
    for (int c = 0; c < C_; ++c)
      cnt += (pat[(c * P_ + p) * N_ + n] == mx) ? 1 : 0;
    const bool act = (sm > 0.0f);
    maxd[p * N_ + n] = act ? mx : __int_as_float(0x7FC00000);
    if (act) atomicAdd(&pcnt[n], cnt);
  }
  __syncthreads();

  for (int n = tid; n < N_; n += 256) psum[n] = pcnt[n];

  for (int idx = tid; idx < 16 * 3 * 32 * 8; idx += 256) {
    const int v    = idx & 7;
    const int lane = (idx >> 3) & 31;
    const int q    = (idx >> 8) % 3;
    const int nt   = idx / 768;
    const int lh   = lane >> 4;
    const int n    = nt * 16 + (lane & 15);
    unsigned int w = 0;
#pragma unroll
    for (int j = 0; j < 4; ++j) {
      const int k = 64 * q + ((v >> 2) << 5) + (lh << 4) + ((v & 3) << 2) + j;
      const int p = k >> 5;
      const int c = k & 31;
      const unsigned int bit =
          (pat[(c * P_ + p) * N_ + n] == maxd[p * N_ + n]) ? 1u : 0u;
      w |= bit << (8 * j);
    }
    Bws[idx] = w;
  }
}

__global__ __launch_bounds__(256) void prep_chars(const float* __restrict__ in,
                                                  unsigned char* __restrict__ chars) {
  const int g = blockIdx.x * 256 + threadIdx.x;
  const int t = g & (T_ - 1);
  const int b = g >> 13;
  int ch = 0;
  for (int c = 0; c < C_; ++c) {
    if (in[(b * C_ + c) * T_ + t] > 0.5f) ch = c;
  }
  chars[g] = (unsigned char)ch;
}

__global__ __launch_bounds__(256) void match_wmma(const unsigned char* __restrict__ chars,
                                                  const unsigned int* __restrict__ Bws,
                                                  const int* __restrict__ psum,
                                                  float* __restrict__ out) {
  const int lane = threadIdx.x & 31;
  const int wave = threadIdx.x >> 5;
  const int g    = blockIdx.x * 8 + wave;
  const int b    = g >> 9;
  const int t0   = (g & 511) << 4;
  const int M    = lane & 15;
  const int half = lane >> 4;
  const int ncol = lane & 15;

  int myc[P_];
  const unsigned char* cb = chars + b * T_;
  const int tr = t0 + M;
#pragma unroll
  for (int p = 0; p < P_; ++p) {
    int tt = tr + p;
    tt = tt < T_ ? tt : T_ - 1;
    myc[p] = (int)cb[tt];
  }

  v8i A[3];
#pragma unroll
  for (int q = 0; q < 3; ++q) {
#pragma unroll
    for (int v = 0; v < 8; ++v) {
      unsigned int w = 0;
#pragma unroll
      for (int j = 0; j < 4; ++j) {
        const int k = 64 * q + ((v >> 2) << 5) + (half << 4) + ((v & 3) << 2) + j;
        const int p = k >> 5;
        const int c = k & 31;
        w |= (myc[p] == c ? 1u : 0u) << (8 * j);
      }
      A[q][v] = (int)w;
    }
  }

  float* ob = out + (size_t)b * N_ * T_;
  const int tbase = t0 + (half << 3);

  for (int nt = 0; nt < 16; ++nt) {
    v8i Bm[3];
#pragma unroll
    for (int q = 0; q < 3; ++q) {
      const unsigned int* bp = Bws + ((size_t)(nt * 3 + q) * 32 + lane) * 8;
#pragma unroll
      for (int v = 0; v < 8; ++v) Bm[q][v] = (int)bp[v];
    }

    v8i acc = {0, 0, 0, 0, 0, 0, 0, 0};
    acc = __builtin_amdgcn_wmma_i32_16x16x64_iu8(false, A[0], false, Bm[0], acc, false, false);
    acc = __builtin_amdgcn_wmma_i32_16x16x64_iu8(false, A[1], false, Bm[1], acc, false, false);
    acc = __builtin_amdgcn_wmma_i32_16x16x64_iu8(false, A[2], false, Bm[2], acc, false, false);

    const int ps = psum[nt * 16 + ncol];
    float f[8];
#pragma unroll
    for (int r = 0; r < 8; ++r) {
      const int t  = tbase + r;
      const int cv = (t < TVALID) ? acc[r] : 0;
      f[r] = (cv == ps) ? 1.0f : 0.0f;
    }

    float4* dst = (float4*)(ob + (size_t)(nt * 16 + ncol) * T_ + tbase);
    dst[0] = make_float4(f[0], f[1], f[2], f[3]);
    dst[1] = make_float4(f[4], f[5], f[6], f[7]);
  }
}

extern "C" void kernel_launch(void* const* d_in, const int* in_sizes, int n_in,
                              void* d_out, int out_size, void* d_ws, size_t ws_size,
                              hipStream_t stream) {
  (void)in_sizes; (void)n_in; (void)out_size; (void)ws_size;
  const float* input    = (const float*)d_in[0];
  const float* patterns = (const float*)d_in[1];

  unsigned int*  Bws   = (unsigned int*)d_ws;
  int*           psum  = (int*)((char*)d_ws + 49152);
  unsigned char* chars = (unsigned char*)((char*)d_ws + 50176);
  float*         out   = (float*)d_out;

  prep_patterns<<<1, 256, 0, stream>>>(patterns, Bws, psum);
  prep_chars<<<(B_ * T_) / 256, 256, 0, stream>>>(input, chars);
  match_wmma<<<(B_ * (T_ / 16)) / 8, 256, 0, stream>>>(chars, Bws, psum, out);
}
